// GraphSAGELayer_66005057405019
// MI455X (gfx1250) — matrix operand order repaired and checked by reading, not compiled, not run on hardware
//
#include <hip/hip_runtime.h>

typedef __attribute__((ext_vector_type(16))) _Float16 v16h;
typedef __attribute__((ext_vector_type(8)))  _Float16 v8h;
typedef __attribute__((ext_vector_type(8)))  float    v8f;

#define EDGE_DIM 16
#define LN_EPS 1e-5f

static __device__ __forceinline__ v8f wmma16(v16h a, v16h b, v8f c) {
  return __builtin_amdgcn_wmma_f32_16x16x32_f16(false, a, false, b, (short)0, c, false, false);
}

static __device__ __forceinline__ void load8(const float* p, float* o) {
  float4 u = *(const float4*)p;
  float4 v = *(const float4*)(p + 4);
  o[0] = u.x; o[1] = u.y; o[2] = u.z; o[3] = u.w;
  o[4] = v.x; o[5] = v.y; o[6] = v.z; o[7] = v.w;
}

// A-matrix 16x32 f16 layout: lane L holds row M=L&15; element i holds
// K = (i&7) + 8*(L>>4) + 16*(i>>3)  ->  two groups of 8 consecutive K values.
static __device__ __forceinline__ v16h pack_a(const float* g0, const float* g1) {
  float t0[8], t1[8];
  load8(g0, t0); load8(g1, t1);
  v16h a;
#pragma unroll
  for (int i = 0; i < 8; ++i) { a[i] = (_Float16)t0[i]; a[i + 8] = (_Float16)t1[i]; }
  return a;
}

static __device__ __forceinline__ v16h pack_a_scaled(const float* g0, const float* g1, float sc) {
  float t0[8], t1[8];
  load8(g0, t0); load8(g1, t1);
  v16h a;
#pragma unroll
  for (int i = 0; i < 8; ++i) { a[i] = (_Float16)(t0[i] * sc); a[i + 8] = (_Float16)(t1[i] * sc); }
  return a;
}

// Stage f32 weights W[kRows x 64] (row-major) into LDS in WMMA B-fragment order:
// frag[tile = c*4+nb][lane][i]  with  K = c*32 + 8*(lane>>4) + (i&7) + 16*(i>>3),  N = nb*16 + (lane&15).
// Rows kRows..kTot-1 are zero padding.  Fragment reads become 2x ds_load_b128.
static __device__ __forceinline__ void stage_frags(_Float16* frag, const float* W,
                                                   int kRows, int kTot, int tid) {
  const int nFrag = (kTot >> 5) * 4;
  for (int j = tid; j < nFrag * 512; j += 128) {
    const int tile = j >> 9;
    const int lane = (j >> 4) & 31;
    const int i    = j & 15;
    const int k    = (tile >> 2) * 32 + ((lane >> 4) << 3) + (i & 7) + ((i >> 3) << 4);
    const int n    = (tile & 3) * 16 + (lane & 15);
    frag[j] = (k < kRows) ? (_Float16)W[k * 64 + n] : (_Float16)0.0f;
  }
}

static __device__ __forceinline__ v16h load_frag(const _Float16* frag, int tile, int lane) {
  const v8h* p = (const v8h*)(frag + ((tile << 5) + lane) * 16);
  v8h lo = p[0], hi = p[1];
  v16h b;
#pragma unroll
  for (int i = 0; i < 8; ++i) { b[i] = lo[i]; b[i + 8] = hi[i]; }
  return b;
}

static __device__ __forceinline__ v16h concat_h(v8h lo, v8h hi) {
  v16h a;
#pragma unroll
  for (int i = 0; i < 8; ++i) { a[i] = lo[i]; a[i + 8] = hi[i]; }
  return a;
}

__global__ void zero_f32_kernel(float* __restrict__ p, long n) {
  long i = (long)blockIdx.x * blockDim.x + threadIdx.x;
  if (i < n) p[i] = 0.0f;
}

// ---------------- Edge kernel: messages + scatter-add ----------------
__global__ __launch_bounds__(128) void edge_msg_kernel(
    const float* __restrict__ x, const int* __restrict__ src, const int* __restrict__ dst,
    const float* __restrict__ ef,
    const float* __restrict__ W1, const float* __restrict__ b1,
    const float* __restrict__ W2, const float* __restrict__ b2,
    float* __restrict__ agg, float* __restrict__ cnt, int E, int numTiles)
{
  __shared__ _Float16 fragW1[12 * 512];   // 12 KB, K padded 80->96
  __shared__ _Float16 fragW2[8 * 512];    // 8 KB
  __shared__ _Float16 sH[4 * 16 * 64];    // 8 KB: per-wave h re-layout buffer

  const int tid  = threadIdx.x;
  const int lane = tid & 31;
  const int wave = tid >> 5;
  const int hl   = lane >> 4;
  const int row  = lane & 15;

  stage_frags(fragW1, W1, 80, 96, tid);
  stage_frags(fragW2, W2, 64, 64, tid);
  __syncthreads();

  // Only layer-1 fragments stay resident (96 VGPRs); layer-2 fragments are
  // re-read from LDS each tile to keep the wave under 256 VGPRs.
  v16h B1[3][4];
#pragma unroll
  for (int c = 0; c < 3; ++c)
#pragma unroll
    for (int nb = 0; nb < 4; ++nb) B1[c][nb] = load_frag(fragW1, c * 4 + nb, lane);

  float bias1[4], bias2[4];
#pragma unroll
  for (int nb = 0; nb < 4; ++nb) { bias1[nb] = b1[nb * 16 + row]; bias2[nb] = b2[nb * 16 + row]; }

  _Float16* hBuf = &sH[wave * 16 * 64];

  const int w0 = blockIdx.x * 4 + wave;
  const int ws = gridDim.x * 4;
  for (int t = w0; t < numTiles; t += ws) {
    const int eBase = t * 16;

    // prefetch next tile's streams (global_prefetch_b8; no counters)
    const int tNext = t + ws;
    if (tNext < numTiles) {
      __builtin_prefetch(src + tNext * 16, 0, 0);
      __builtin_prefetch(dst + tNext * 16, 0, 0);
      __builtin_prefetch(ef + (long)(tNext * 16 + row) * EDGE_DIM, 0, 0);
    }

    const int e = eBase + row;
    const bool ev = (e < E);
    const int s = ev ? src[e] : 0;
    const float* xr = x + (long)s * 64;
    const float* er = ef + (long)(ev ? e : 0) * EDGE_DIM;

    // A (16 x 96): [x_src(64) | ef(16) | 0(16)]
    v16h a0 = pack_a(xr + 8 * hl,      xr + 16 + 8 * hl);
    v16h a1 = pack_a(xr + 32 + 8 * hl, xr + 48 + 8 * hl);
    v16h a2;
#pragma unroll
    for (int i = 0; i < 8; ++i) { a2[i] = (_Float16)er[8 * hl + i]; a2[i + 8] = (_Float16)0.0f; }

    // layer 1: h = relu(A @ W1 + b1)
    v8f acc1[4];
#pragma unroll
    for (int nb = 0; nb < 4; ++nb) {
      v8f c;
#pragma unroll
      for (int r = 0; r < 8; ++r) c[r] = bias1[nb];
      c = wmma16(a0, B1[0][nb], c);
      c = wmma16(a1, B1[1][nb], c);
      c = wmma16(a2, B1[2][nb], c);
      acc1[nb] = c;
    }

    // relu + D-layout -> row-major re-layout through per-wave LDS (in-order per wave)
#pragma unroll
    for (int nb = 0; nb < 4; ++nb)
#pragma unroll
      for (int r = 0; r < 8; ++r) {
        float v = acc1[nb][r];
        v = v > 0.0f ? v : 0.0f;
        hBuf[(r + 8 * hl) * 64 + nb * 16 + row] = (_Float16)v;
      }

    // compile-time fence: keeps the sH stores before the reads below and
    // stops LICM from hoisting the fragW2 reads (register-pressure control)
    asm volatile("" ::: "memory");

    const _Float16* hRow = hBuf + row * 64;
    const v8h* hv = (const v8h*)(hRow + 8 * hl);        // 16B aligned
    v16h h0 = concat_h(hv[0], hv[2]);                   // K 0..31   (this lane's halves)
    v16h h1 = concat_h(hv[4], hv[6]);                   // K 32..63

    // layer 2: messages = h @ W2 + b2   (B fragments streamed from LDS)
    v8f acc2[4];
#pragma unroll
    for (int nb = 0; nb < 4; ++nb) {
      v8f c;
#pragma unroll
      for (int r = 0; r < 8; ++r) c[r] = bias2[nb];
      c = wmma16(h0, load_frag(fragW2, nb,     lane), c);
      c = wmma16(h1, load_frag(fragW2, 4 + nb, lane), c);
      acc2[nb] = c;
    }

    // scatter-add: D element (r,lane) = message row (r+8*hl), col nb*16+row
#pragma unroll
    for (int r = 0; r < 8; ++r) {
      const int eR = eBase + r + 8 * hl;
      if (eR < E) {
        const int d = dst[eR];
        float* ap = agg + (long)d * 64 + row;
#pragma unroll
        for (int nb = 0; nb < 4; ++nb)
          unsafeAtomicAdd(ap + nb * 16, acc2[nb][r]);
      }
    }
    if (lane < 16) {
      const int ee = eBase + lane;
      if (ee < E) unsafeAtomicAdd(&cnt[dst[ee]], 1.0f);
    }
  }
}

// ---------------- Node kernel: update MLP + residual + LayerNorm ----------------
__global__ __launch_bounds__(128) void node_update_kernel(
    const float* __restrict__ x, const float* __restrict__ agg, const float* __restrict__ cnt,
    const float* __restrict__ W3, const float* __restrict__ b3,
    const float* __restrict__ gamma, const float* __restrict__ beta,
    float* __restrict__ out, int N, int numTiles)
{
  __shared__ _Float16 fragW3[16 * 512];   // 16 KB

  const int tid  = threadIdx.x;
  const int lane = tid & 31;
  const int wave = tid >> 5;
  const int hl   = lane >> 4;
  const int row  = lane & 15;

  stage_frags(fragW3, W3, 128, 128, tid);
  __syncthreads();

  v16h B3[4][4];
#pragma unroll
  for (int c = 0; c < 4; ++c)
#pragma unroll
    for (int nb = 0; nb < 4; ++nb) B3[c][nb] = load_frag(fragW3, c * 4 + nb, lane);

  float bias3[4], gam[4], bet[4];
#pragma unroll
  for (int nb = 0; nb < 4; ++nb) {
    bias3[nb] = b3[nb * 16 + row];
    gam[nb]   = gamma[nb * 16 + row];
    bet[nb]   = beta[nb * 16 + row];
  }

  const int w0 = blockIdx.x * 4 + wave;
  const int ws = gridDim.x * 4;
  for (int t = w0; t < numTiles; t += ws) {
    const int tNext = t + ws;
    if (tNext < numTiles) {
      __builtin_prefetch(x   + (long)(tNext * 16 + row) * 64, 0, 0);
      __builtin_prefetch(agg + (long)(tNext * 16 + row) * 64, 0, 0);
      __builtin_prefetch(cnt + tNext * 16, 0, 0);
    }

    const int node = t * 16 + row;
    const int nn = node < N ? node : 0;
    const float* xr = x   + (long)nn * 64;
    const float* ar = agg + (long)nn * 64;
    const float inv = 1.0f / (cnt[nn] + 1e-8f);

    // A (16 x 128): [x(64) | agg/deg(64)]
    v16h a0 = pack_a(xr + 8 * hl,      xr + 16 + 8 * hl);
    v16h a1 = pack_a(xr + 32 + 8 * hl, xr + 48 + 8 * hl);
    v16h a2 = pack_a_scaled(ar + 8 * hl,      ar + 16 + 8 * hl, inv);
    v16h a3 = pack_a_scaled(ar + 32 + 8 * hl, ar + 48 + 8 * hl, inv);

    v8f acc[4];
#pragma unroll
    for (int nb = 0; nb < 4; ++nb) {
      v8f c;
#pragma unroll
      for (int r = 0; r < 8; ++r) c[r] = bias3[nb];
      c = wmma16(a0, B3[0][nb], c);
      c = wmma16(a1, B3[1][nb], c);
      c = wmma16(a2, B3[2][nb], c);
      c = wmma16(a3, B3[3][nb], c);
      acc[nb] = c;
    }

    // y = relu(acc) + x ; LayerNorm over 64 cols per row (row lives on 16 lanes)
#pragma unroll
    for (int r = 0; r < 8; ++r) {
      const int nodeR = t * 16 + r + 8 * hl;
      const int nr = nodeR < N ? nodeR : 0;
      float y[4];
      float p1 = 0.0f, p2 = 0.0f;
#pragma unroll
      for (int nb = 0; nb < 4; ++nb) {
        float v = acc[nb][r];
        v = v > 0.0f ? v : 0.0f;
        v += x[(long)nr * 64 + nb * 16 + row];
        y[nb] = v;
        p1 += v; p2 += v * v;
      }
      // reduce across the 16 lanes of this half-wave (xor masks 1,2,4,8 stay in-group)
#pragma unroll
      for (int m = 1; m < 16; m <<= 1) {
        p1 += __shfl_xor(p1, m, 32);
        p2 += __shfl_xor(p2, m, 32);
      }
      const float mu  = p1 * (1.0f / 64.0f);
      const float var = p2 * (1.0f / 64.0f) - mu * mu;
      const float rs  = rsqrtf(var + LN_EPS);
      if (nodeR < N) {
#pragma unroll
        for (int nb = 0; nb < 4; ++nb)
          out[(long)nodeR * 64 + nb * 16 + row] = (y[nb] - mu) * rs * gam[nb] + bet[nb];
      }
    }
  }
}

extern "C" void kernel_launch(void* const* d_in, const int* in_sizes, int n_in,
                              void* d_out, int out_size, void* d_ws, size_t ws_size,
                              hipStream_t stream) {
  const float* x     = (const float*)d_in[0];
  const int*   ei    = (const int*)d_in[1];
  const float* ef    = (const float*)d_in[2];
  const float* W1    = (const float*)d_in[3];
  const float* b1    = (const float*)d_in[4];
  const float* W2    = (const float*)d_in[5];
  const float* b2    = (const float*)d_in[6];
  const float* W3    = (const float*)d_in[7];
  const float* b3    = (const float*)d_in[8];
  const float* gamma = (const float*)d_in[9];
  const float* beta  = (const float*)d_in[10];

  const int N = in_sizes[0] / 64;
  const int E = in_sizes[1] / 2;
  const int* src = ei;
  const int* dst = ei + E;

  float* agg = (float*)d_ws;              // N*64 f32
  float* cnt = agg + (long)N * 64;        // N f32

  const long zn = (long)N * 64 + N;
  zero_f32_kernel<<<(unsigned)((zn + 255) / 256), 256, 0, stream>>>(agg, zn);

  const int eTiles = (E + 15) / 16;
  int eBlocks = (eTiles + 3) / 4; if (eBlocks > 1280) eBlocks = 1280;
  edge_msg_kernel<<<eBlocks, 128, 0, stream>>>(x, src, dst, ef, W1, b1, W2, b2,
                                               agg, cnt, E, eTiles);

  const int nTiles = (N + 15) / 16;
  int nBlocks = (nTiles + 3) / 4; if (nBlocks > 1280) nBlocks = 1280;
  node_update_kernel<<<nBlocks, 128, 0, stream>>>(x, agg, cnt, W3, b3, gamma, beta,
                                                  (float*)d_out, N, nTiles);
}
